// _CTRTokenMixer_89532888253114
// MI455X (gfx1250) — matrix operand order repaired and checked by reading, not compiled, not run on hardware
//
#include <hip/hip_runtime.h>
#include <hip/hip_bf16.h>
#include <math.h>

// ---------------------------------------------------------------------------
// CTR token mixer for gfx1250 (MI455X).
//   T=4096 (64x64), B=16, C=256, H=8, D=32, M=16 patches, TOPK=2, SIM_POW=2
// Three 65536x256x256 GEMMs done with v_wmma_f32_16x16x32_f16; clustering
// stage done per (b,h) workgroup with LDS f32 atomics.
// ---------------------------------------------------------------------------

typedef _Float16 h8  __attribute__((ext_vector_type(8)));
typedef _Float16 h4  __attribute__((ext_vector_type(4)));
typedef _Float16 v16h __attribute__((ext_vector_type(16)));
typedef float    v8f __attribute__((ext_vector_type(8)));
typedef float    f4  __attribute__((ext_vector_type(4)));

#define TOKS   4096
#define BATCH  16
#define CDIM   256
#define NROWS  (TOKS * BATCH)   // 65536 GEMM rows, row n = t*B + b

// ---------------------------------------------------------------------------
// fp32 -> fp16 convert, 4 elems/thread
// ---------------------------------------------------------------------------
__global__ __launch_bounds__(256) void cvt_f32_f16(const float* __restrict__ s,
                                                   _Float16* __restrict__ d,
                                                   int n4) {
  int i = blockIdx.x * 256 + threadIdx.x;
  if (i < n4) {
    f4 x = ((const f4*)s)[i];
    h4 y;
    y[0] = (_Float16)x[0];
    y[1] = (_Float16)x[1];
    y[2] = (_Float16)x[2];
    y[3] = (_Float16)x[3];
    ((h4*)d)[i] = y;
  }
}

// ---------------------------------------------------------------------------
// GEMM: out[n, c] = sum_k A[n,k] * W[c,k] + bias[c]
//   A: [rows, 256] f16 row-major, W: [256, 256] f16 row-major (W[c][k])
// Workgroup = 256 threads = 8 waves, tile 128(M) x 128(N), BK=32.
// Wave (wm in 0..1, wn in 0..3) owns 64x32 = 4x2 tiles of 16x16.
// LDS rows padded to 40 halves (80 B) so every fragment chunk is a 16 B load.
// ---------------------------------------------------------------------------
template <typename OutT>
__global__ __launch_bounds__(256) void gemm256(const _Float16* __restrict__ A,
                                               const _Float16* __restrict__ W,
                                               const float* __restrict__ bias,
                                               OutT* __restrict__ out) {
  __shared__ _Float16 As[128 * 40];
  __shared__ _Float16 Bs[128 * 40];

  const int tid  = threadIdx.x;
  const int lane = tid & 31;
  const int wid  = tid >> 5;
  const int wm   = wid >> 2;   // 0..1
  const int wn   = wid & 3;    // 0..3
  const int lg   = lane >> 4;  // 0/1: K-half group
  const int lr   = lane & 15;  // row (A) / col (B) within 16

  const int rowBase = blockIdx.x * 128;
  const int colBase = blockIdx.y * 128;

  v8f acc[4][2] = {};

  for (int kb = 0; kb < 8; ++kb) {
    const int k0 = kb * 32;
    __syncthreads();
    // Stage A tile (128 rows x 32 k) and W tile (128 cols x 32 k), 8 halves/thr/pass.
#pragma unroll
    for (int p = 0; p < 2; ++p) {
      int idx = p * 256 + tid;        // 0..511
      int r   = idx >> 2;             // 0..127
      int g   = (idx & 3) * 8;        // 0,8,16,24
      *(h8*)&As[r * 40 + g] = *(const h8*)&A[(size_t)(rowBase + r) * CDIM + k0 + g];
      *(h8*)&Bs[r * 40 + g] = *(const h8*)&W[(size_t)(colBase + r) * CDIM + k0 + g];
    }
    __syncthreads();

    // B fragments: lane lr = column, halves j -> K = lg*16 + j
    v16h bf[2];
#pragma unroll
    for (int nt = 0; nt < 2; ++nt) {
      int c = wn * 32 + nt * 16 + lr;
      const h8* bp = (const h8*)&Bs[c * 40 + lg * 16];
      h8 b0 = bp[0], b1 = bp[1];
      bf[nt] = __builtin_shufflevector(b0, b1, 0, 1, 2, 3, 4, 5, 6, 7,
                                       8, 9, 10, 11, 12, 13, 14, 15);
    }
    // A fragments: lane lr = row; halves j -> K = lg*16 + j
#pragma unroll
    for (int mt = 0; mt < 4; ++mt) {
      int r = wm * 64 + mt * 16 + lr;
      h8 a0 = *(const h8*)&As[r * 40 + lg * 16];
      h8 a1 = *(const h8*)&As[r * 40 + lg * 16 + 8];
      v16h af = __builtin_shufflevector(a0, a1, 0, 1, 2, 3, 4, 5, 6, 7,
                                        8, 9, 10, 11, 12, 13, 14, 15);
      acc[mt][0] = __builtin_amdgcn_wmma_f32_16x16x32_f16(
          false, af, false, bf[0], (short)0, acc[mt][0], false, false);
      acc[mt][1] = __builtin_amdgcn_wmma_f32_16x16x32_f16(
          false, af, false, bf[1], (short)0, acc[mt][1], false, false);
    }
  }

  // Epilogue: D layout lane lr = col, vgpr i -> row = lg*8 + i.
#pragma unroll
  for (int mt = 0; mt < 4; ++mt) {
#pragma unroll
    for (int nt = 0; nt < 2; ++nt) {
      int col  = colBase + wn * 32 + nt * 16 + lr;
      float bv = bias[col];
      int rb   = rowBase + wm * 64 + mt * 16 + lg * 8;
#pragma unroll
      for (int i = 0; i < 8; ++i) {
        out[(size_t)(rb + i) * CDIM + col] = (OutT)(acc[mt][nt][i] + bv);
      }
    }
  }
}

// ---------------------------------------------------------------------------
// Clustering stage: one workgroup per (b,h).  q,v: [n=t*B+b, h*32+d] f16.
// Writes merged out_tok y (same layout, f16).
// ---------------------------------------------------------------------------
__global__ __launch_bounds__(256) void stage_cluster(
    const _Float16* __restrict__ q, const _Float16* __restrict__ v,
    const float* __restrict__ alpha_p, const float* __restrict__ beta_p,
    _Float16* __restrict__ y) {
  __shared__ float cent[16][32];   // centers (normalized), later centers_out
  __shared__ float vcent[16][32];
  __shared__ float csum[16][32];
  __shared__ float ssum[16];
  __shared__ float wgt[2][4096];
  __shared__ unsigned pidx[4096];

  const int bh  = blockIdx.x;
  const int b   = bh >> 3;
  const int h   = bh & 7;
  const int tid = threadIdx.x;
  const float alpha = alpha_p[0];
  const float beta  = beta_p[0];
  const size_t colOff = (size_t)h * 32;

  // --- adaptive avg pool 64x64 -> 4x4 (m = pw*4 + ph), for q and v ---
#pragma unroll
  for (int p = 0; p < 2; ++p) {
    int vi = tid + p * 256;      // 0..511 -> (m, d)
    int m = vi >> 5, d = vi & 31;
    int pw = m >> 2, ph = m & 3;
    float sq = 0.0f, sv = 0.0f;
    for (int yy = 0; yy < 16; ++yy) {
      int t0 = (pw * 16 + yy) * 64 + ph * 16;
      for (int xx = 0; xx < 16; ++xx) {
        size_t off = ((size_t)(t0 + xx) * BATCH + b) * CDIM + colOff + d;
        sq += (float)q[off];
        sv += (float)v[off];
      }
    }
    cent[m][d]  = sq * (1.0f / 256.0f);
    vcent[m][d] = sv * (1.0f / 256.0f);
    csum[m][d]  = 0.0f;
  }
  if (tid < 16) ssum[tid] = 0.0f;
  __syncthreads();

  // --- L2-normalize centers ---
  if (tid < 16) {
    float s = 0.0f;
    for (int d = 0; d < 32; ++d) s += cent[tid][d] * cent[tid][d];
    float rn = 1.0f / fmaxf(sqrtf(s), 1e-12f);
    for (int d = 0; d < 32; ++d) cent[tid][d] *= rn;
  }
  __syncthreads();

  // --- per-token: sim, top-2, ^2-normalize, scatter w*v into centers ---
  for (int tt = 0; tt < 16; ++tt) {
    int t = tt * 256 + tid;
    size_t off = ((size_t)t * BATCH + b) * CDIM + colOff;
    float qd[32];
    const h8* qp = (const h8*)(q + off);
#pragma unroll
    for (int c = 0; c < 4; ++c) {
      h8 hv = qp[c];
#pragma unroll
      for (int j = 0; j < 8; ++j) qd[c * 8 + j] = (float)hv[j];
    }
    float s = 0.0f;
#pragma unroll
    for (int d = 0; d < 32; ++d) s += qd[d] * qd[d];
    float rn = 1.0f / fmaxf(sqrtf(s), 1e-12f);

    float b1 = -1.0f, b2 = -1.0f;
    int i1 = 0, i2 = 0;
    for (int m = 0; m < 16; ++m) {
      float dot = 0.0f;
#pragma unroll
      for (int d = 0; d < 32; ++d) dot += cent[m][d] * qd[d];
      float z  = beta + alpha * dot * rn;
      float sg = 1.0f / (1.0f + __expf(-z));
      if (sg > b1) { b2 = b1; i2 = i1; b1 = sg; i1 = m; }
      else if (sg > b2) { b2 = sg; i2 = m; }
    }
    float w1 = b1 * b1, w2 = b2 * b2;           // SIM_POW = 2
    float inv = 1.0f / (w1 + w2 + 1e-6f);
    w1 *= inv; w2 *= inv;
    wgt[0][t] = w1;
    wgt[1][t] = w2;
    pidx[t] = (unsigned)i1 | ((unsigned)i2 << 8);

    const h8* vp = (const h8*)(v + off);
    float vd[32];
#pragma unroll
    for (int c = 0; c < 4; ++c) {
      h8 hv = vp[c];
#pragma unroll
      for (int j = 0; j < 8; ++j) vd[c * 8 + j] = (float)hv[j];
    }
#pragma unroll
    for (int d = 0; d < 32; ++d) atomicAdd(&csum[i1][d], w1 * vd[d]);
#pragma unroll
    for (int d = 0; d < 32; ++d) atomicAdd(&csum[i2][d], w2 * vd[d]);
    atomicAdd(&ssum[i1], w1);
    atomicAdd(&ssum[i2], w2);
  }
  __syncthreads();

  // --- centers_out = (csum + value_centers) / (ssum + 1) ---
#pragma unroll
  for (int p = 0; p < 2; ++p) {
    int vi = tid + p * 256;
    int m = vi >> 5, d = vi & 31;
    cent[m][d] = (csum[m][d] + vcent[m][d]) / (ssum[m] + 1.0f);
  }
  __syncthreads();

  // --- out_tok[t] = w1*co[i1] + w2*co[i2] -> merged f16 y ---
  for (int tt = 0; tt < 16; ++tt) {
    int t = tt * 256 + tid;
    float w1 = wgt[0][t], w2 = wgt[1][t];
    unsigned pk = pidx[t];
    int i1 = pk & 255, i2 = (pk >> 8) & 255;
    size_t off = ((size_t)t * BATCH + b) * CDIM + colOff;
    h8* yp = (h8*)(y + off);
#pragma unroll
    for (int c = 0; c < 4; ++c) {
      h8 o;
#pragma unroll
      for (int j = 0; j < 8; ++j) {
        int d = c * 8 + j;
        o[j] = (_Float16)(w1 * cent[i1][d] + w2 * cent[i2][d]);
      }
      yp[c] = o;
    }
  }
}

// ---------------------------------------------------------------------------
// Host launch
// ---------------------------------------------------------------------------
extern "C" void kernel_launch(void* const* d_in, const int* in_sizes, int n_in,
                              void* d_out, int out_size, void* d_ws, size_t ws_size,
                              hipStream_t stream) {
  const float* x      = (const float*)d_in[0];
  const float* f_w    = (const float*)d_in[1];
  const float* f_b    = (const float*)d_in[2];
  const float* v_w    = (const float*)d_in[3];
  const float* v_b    = (const float*)d_in[4];
  const float* proj_w = (const float*)d_in[5];
  const float* proj_b = (const float*)d_in[6];
  const float* alpha  = (const float*)d_in[7];
  const float* beta   = (const float*)d_in[8];

  char* ws = (char*)d_ws;
  const size_t MB = 1u << 20;
  _Float16* xh  = (_Float16*)(ws);              // 32 MB, reused as y after GEMMs
  _Float16* qh  = (_Float16*)(ws + 32 * MB);    // 32 MB
  _Float16* vh  = (_Float16*)(ws + 64 * MB);    // 32 MB
  _Float16* fwh = (_Float16*)(ws + 96 * MB);    // 128 KB
  _Float16* vwh = fwh + CDIM * CDIM;
  _Float16* pwh = vwh + CDIM * CDIM;

  // fp32 -> fp16
  cvt_f32_f16<<<(NROWS * CDIM / 4 + 255) / 256, 256, 0, stream>>>(x, xh, NROWS * CDIM / 4);
  cvt_f32_f16<<<64, 256, 0, stream>>>(f_w, fwh, CDIM * CDIM / 4);
  cvt_f32_f16<<<64, 256, 0, stream>>>(v_w, vwh, CDIM * CDIM / 4);
  cvt_f32_f16<<<64, 256, 0, stream>>>(proj_w, pwh, CDIM * CDIM / 4);

  dim3 g(NROWS / 128, CDIM / 128);  // 512 x 2
  gemm256<_Float16><<<g, 256, 0, stream>>>(xh, fwh, f_b, qh);
  gemm256<_Float16><<<g, 256, 0, stream>>>(xh, vwh, v_b, vh);

  stage_cluster<<<128, 256, 0, stream>>>(qh, vh, alpha, beta, xh /* y overwrites xh */);

  // proj: rows are n = t*B + b, which is exactly d_out's [T,B,C] layout
  gemm256<float><<<g, 256, 0, stream>>>(xh, pwh, proj_b, (float*)d_out);
}
